// VoxelResBackBone8xImpNotImpGen_56624848831235
// MI455X (gfx1250) — compile-verified, matrix operand order checked by reading, not run on hardware
//
#include <hip/hip_runtime.h>

typedef __attribute__((ext_vector_type(16))) __bf16 v16bf;
typedef __attribute__((ext_vector_type(8)))  float  v8f;
typedef unsigned short u16;
typedef unsigned int   u32;

// ---------- bf16 helpers (round-to-nearest-even) ----------
__device__ __forceinline__ u16 f2bf(float f) {
    u32 u = __float_as_uint(f);
    u32 r = (u + 0x7FFFu + ((u >> 16) & 1u)) >> 16;
    return (u16)r;
}
__device__ __forceinline__ float bf2f(u16 h) {
    return __uint_as_float(((u32)h) << 16);
}

union Frag {
    v16bf v;
    uint4 u[2];
};

// ---------- utility kernels ----------
__global__ void k_zero16(uint4* p, long n16) {
    long i = (long)blockIdx.x * blockDim.x + threadIdx.x;
    if (i < n16) p[i] = make_uint4(0u, 0u, 0u, 0u);
}

__global__ void k_scatter(const float* __restrict__ feat, const int* __restrict__ coords,
                          int N, u16* __restrict__ grid, u16* __restrict__ mask,
                          int H, int W, int Cpad) {
    int i = blockIdx.x * blockDim.x + threadIdx.x;
    if (i >= N) return;
    int z = coords[i * 4 + 1], y = coords[i * 4 + 2], x = coords[i * 4 + 3];
    long v = ((long)z * H + y) * W + x;
    long base = v * Cpad;
    grid[base + 0] = f2bf(feat[i * 4 + 0]);
    grid[base + 1] = f2bf(feat[i * 4 + 1]);
    grid[base + 2] = f2bf(feat[i * 4 + 2]);
    grid[base + 3] = f2bf(feat[i * 4 + 3]);
    mask[v] = 0x3F80; // bf16 1.0
}

// pack fp32 weights [taps][cin][cout] -> bf16 Wt[cout][KtotPad]
// (channel-padded to cinPad, K-padded with zeros to a multiple of 32)
__global__ void k_packw(const float* __restrict__ w, u16* __restrict__ wt,
                        int taps, int cin, int cinPad, int cout, int KtotPad) {
    long total = (long)cout * KtotPad;
    long i = (long)blockIdx.x * blockDim.x + threadIdx.x;
    if (i >= total) return;
    int kk = (int)(i % KtotPad);
    int n  = (int)(i / KtotPad);
    int tap = kk / cinPad;
    int c   = kk % cinPad;
    float val = (tap < taps && c < cin) ? w[((long)tap * cin + c) * cout + n] : 0.0f;
    wt[i] = f2bf(val);
}

// fold conv-bias + BN into per-channel scale/shift
__global__ void k_bnfold(const float* __restrict__ g, const float* __restrict__ b,
                         const float* __restrict__ m, const float* __restrict__ v,
                         const float* __restrict__ bias, int C,
                         float* __restrict__ scale, float* __restrict__ shift) {
    int c = blockIdx.x * blockDim.x + threadIdx.x;
    if (c >= C) return;
    float s = g[c] * rsqrtf(v[c] + 1.0e-3f);
    scale[c] = s;
    float bi = bias ? bias[c] : 0.0f;
    shift[c] = (bi - m[c]) * s + b[c];
}

// mask downsample: OR over kernel window, write bf16 1.0/0.0
__global__ void k_maskdown(const u16* __restrict__ mi, u16* __restrict__ mo,
                           int Din, int Hin, int Win, int Dout, int Hout, int Wout,
                           int kd, int kh, int kw, int sd, int sh, int sw,
                           int pd, int ph, int pw) {
    long tot = (long)Dout * Hout * Wout;
    long i = (long)blockIdx.x * blockDim.x + threadIdx.x;
    if (i >= tot) return;
    int ow = (int)(i % Wout);
    long t = i / Wout;
    int oh = (int)(t % Hout);
    int od = (int)(t / Hout);
    int any = 0;
    for (int a = 0; a < kd; ++a) {
        int id = od * sd - pd + a;
        if (id < 0 || id >= Din) continue;
        for (int b2 = 0; b2 < kh; ++b2) {
            int ih = oh * sh - ph + b2;
            if (ih < 0 || ih >= Hin) continue;
            for (int c = 0; c < kw; ++c) {
                int iw = ow * sw - pw + c;
                if (iw < 0 || iw >= Win) continue;
                if (mi[((long)id * Hin + ih) * Win + iw]) any = 1;
            }
        }
    }
    mo[i] = any ? (u16)0x3F80 : (u16)0;
}

// ---------- implicit-GEMM conv with WMMA bf16 ----------
struct ConvArgs {
    const u16* __restrict__ in;    // [Din,Hin,Win,CIN] bf16
    const u16* __restrict__ wt;    // [Cout][KtotPad] bf16 (transposed, zero-padded)
    const float* __restrict__ scale;
    const float* __restrict__ shift;
    const u16* __restrict__ mask;  // [Mtot] bf16 or null
    const u16* __restrict__ res;   // [Mtot,Cout] bf16 or null
    const u16* __restrict__ zero;  // 64B zero page for OOB A loads
    u16* __restrict__ out_bf;      // bf16 output or null
    float* __restrict__ out_f;     // fp32 output or null
    int Din, Hin, Win;
    int Dout, Hout, Wout, Cout;
    int sd, sh, sw, pd, ph, pw;
    int relu, Mtot, ntiles, totalWaves, Ktot; // Ktot == KtotPad (multiple of 32)
};

// CIN: padded input channels (16/32/64/128); KW,KH: kernel extents (3,3 or 1,1);
// NT: N-tiles (16 channels each) computed per wave.
template <int CIN, int KW, int KH, int NT>
__global__ __launch_bounds__(128) void k_conv_wmma(ConvArgs p) {
    constexpr int L2C = (CIN == 16) ? 4 : (CIN == 32) ? 5 : (CIN == 64) ? 6 : 7;
    int wid = blockIdx.x * 4 + (threadIdx.x >> 5);
    if (wid >= p.totalWaves) return; // uniform per-wave: EXEC all-ones for WMMA
    int lane = threadIdx.x & 31;
    int nblocks = p.ntiles / NT;
    int mtile = wid / nblocks;
    int nb = wid - mtile * nblocks;
    int ml = lane & 15;  // M row (A) / N col (B,C)
    int h  = lane >> 4;  // K-half selector per ISA 16-bit fragment layout
    int m = mtile * 16 + ml;
    bool mvalid = m < p.Mtot;
    int mm = mvalid ? m : 0;
    int ow = mm % p.Wout;
    int tq = mm / p.Wout;
    int oh = tq % p.Hout;
    int od = tq / p.Hout;
    int idb = od * p.sd - p.pd, ihb = oh * p.sh - p.ph, iwb = ow * p.sw - p.pw;

    // Precompute per-lane tap validity bitmask (taps <= 28 incl. K-padding).
    int taps = p.Ktot >> L2C;
    u32 vmask = 0;
    for (int t = 0; t < taps; ++t) {
        int kw_ = t % KW;
        int t2 = t / KW;
        int kh_ = t2 % KH;
        int kd_ = t2 / KH;
        int id = idb + kd_, ih = ihb + kh_, iw = iwb + kw_;
        if (mvalid && id >= 0 && id < p.Din && ih >= 0 && ih < p.Hin &&
            iw >= 0 && iw < p.Win)
            vmask |= (1u << t);
    }

    int n0 = nb * (NT * 16) + ml;
    const u16* wrow[NT];
#pragma unroll
    for (int j = 0; j < NT; ++j) wrow[j] = p.wt + (long)(n0 + j * 16) * p.Ktot;

    v8f acc[NT] = {};

    for (int kb = 0; kb < p.Ktot; kb += 32) {
        int k0 = kb + h * 8;       // this lane's two 8-element K runs
        int k1 = kb + 16 + h * 8;
        Frag A;
#pragma unroll
        for (int part = 0; part < 2; ++part) {
            // Wave-uniform tap index (scalar path), per-lane channel offset.
            int tap, c;
            if constexpr (CIN == 16) {
                tap = (kb >> 4) + part;     // h*8 < 16: uniform across lanes
                c = h * 8;
            } else {
                tap = kb >> L2C;            // part*16 + h*8 < CIN: uniform
                c = (kb & (CIN - 1)) + part * 16 + h * 8;
            }
            int kw_ = tap % KW;             // scalar constant-divisor math
            int t2 = tap / KW;
            int kh_ = t2 % KH;
            int kd_ = t2 / KH;
            long offs = (((long)(idb + kd_) * p.Hin + (ihb + kh_)) * p.Win +
                         (iwb + kw_)) * CIN + c;
            // Branchless: select real address or 64B zero page.
            const u16* pa = ((vmask >> tap) & 1u) ? (p.in + offs) : p.zero;
            A.u[part] = *(const uint4*)pa;
        }
#pragma unroll
        for (int j = 0; j < NT; ++j) {
            Frag B;
            B.u[0] = *(const uint4*)(wrow[j] + k0);
            B.u[1] = *(const uint4*)(wrow[j] + k1);
            acc[j] = __builtin_amdgcn_wmma_f32_16x16x32_bf16(
                false, A.v, false, B.v, (short)0, acc[j], false, false);
        }
    }

#pragma unroll
    for (int j = 0; j < NT; ++j) {
        int n = n0 + j * 16;
        float sc = p.scale[n], sh0 = p.shift[n];
#pragma unroll
        for (int r = 0; r < 8; ++r) {
            int mo = mtile * 16 + h * 8 + r; // C/D: lanes 0-15 M=r, lanes 16-31 M=r+8
            if (mo < p.Mtot) {
                float y = acc[j][r] * sc + sh0;
                long o = (long)mo * p.Cout + n;
                if (p.res)  y += bf2f(p.res[o]);
                if (p.relu) y = fmaxf(y, 0.0f);
                if (p.mask) y *= bf2f(p.mask[mo]);
                if (p.out_bf) p.out_bf[o] = f2bf(y);
                else          p.out_f[o]  = y;
            }
        }
    }
}

// ---------- host-side orchestration ----------
struct Bump {
    char* base; size_t off;
    void* alloc(size_t bytes) {
        size_t a = (off + 255) & ~(size_t)255;
        off = a + bytes;
        return base + a;
    }
};

struct PC { u16* wt; float* sc; float* sh; int taps, cin, cinPad, cout, ktotPad; };

static PC make_pc(Bump& ws, int taps, int cin, int cinPad, int cout) {
    PC p; p.taps = taps; p.cin = cin; p.cinPad = cinPad; p.cout = cout;
    p.ktotPad = (taps * cinPad + 31) & ~31;
    p.wt = (u16*)ws.alloc((size_t)cout * p.ktotPad * sizeof(u16));
    p.sc = (float*)ws.alloc((size_t)cout * sizeof(float));
    p.sh = (float*)ws.alloc((size_t)cout * sizeof(float));
    return p;
}

static void prep_pc(const PC& p, const float* w, const float* bias,
                    const float* bnb, const float* bng, const float* bnm, const float* bnv,
                    hipStream_t s) {
    long tot = (long)p.cout * p.ktotPad;
    k_packw<<<(int)((tot + 255) / 256), 256, 0, s>>>(w, p.wt, p.taps, p.cin, p.cinPad,
                                                    p.cout, p.ktotPad);
    k_bnfold<<<(p.cout + 255) / 256, 256, 0, s>>>(bng, bnb, bnm, bnv, bias, p.cout, p.sc, p.sh);
}

static void run_conv(const u16* in, const PC& pc, const u16* mask, const u16* res,
                     const u16* zero, u16* outb, float* outf,
                     int Din, int Hin, int Win, int Dout, int Hout, int Wout,
                     int kd, int kh, int kw, int sd, int sh, int sw,
                     int pd, int ph, int pw, int relu, hipStream_t s) {
    ConvArgs a{};
    a.in = in; a.wt = pc.wt; a.scale = pc.sc; a.shift = pc.sh;
    a.mask = mask; a.res = res; a.zero = zero; a.out_bf = outb; a.out_f = outf;
    a.Din = Din; a.Hin = Hin; a.Win = Win;
    a.Dout = Dout; a.Hout = Hout; a.Wout = Wout; a.Cout = pc.cout;
    a.sd = sd; a.sh = sh; a.sw = sw; a.pd = pd; a.ph = ph; a.pw = pw;
    a.relu = relu;
    a.Mtot = Dout * Hout * Wout;
    a.ntiles = pc.cout / 16;
    a.Ktot = pc.ktotPad;
    int nt = a.ntiles >= 4 ? 4 : a.ntiles;
    int mtiles = (a.Mtot + 15) / 16;
    a.totalWaves = mtiles * (a.ntiles / nt);
    int blocks = (a.totalWaves + 3) / 4;
    bool k3 = (kw == 3);
    if (pc.cinPad == 16) {
        if (nt == 1) k_conv_wmma<16, 3, 3, 1><<<blocks, 128, 0, s>>>(a);
        else         k_conv_wmma<16, 3, 3, 2><<<blocks, 128, 0, s>>>(a);
    } else if (pc.cinPad == 32) {
        if (nt == 2) k_conv_wmma<32, 3, 3, 2><<<blocks, 128, 0, s>>>(a);
        else         k_conv_wmma<32, 3, 3, 4><<<blocks, 128, 0, s>>>(a);
    } else if (pc.cinPad == 64) {
        k_conv_wmma<64, 3, 3, 4><<<blocks, 128, 0, s>>>(a);
    } else {
        if (k3) k_conv_wmma<128, 3, 3, 4><<<blocks, 128, 0, s>>>(a);
        else    k_conv_wmma<128, 1, 1, 4><<<blocks, 128, 0, s>>>(a);
    }
}

extern "C" void kernel_launch(void* const* d_in, const int* in_sizes, int n_in,
                              void* d_out, int out_size, void* d_ws, size_t ws_size,
                              hipStream_t stream) {
    const float* feat = (const float*)d_in[0];
    const int* coords = (const int*)d_in[1];
    int N = in_sizes[0] / 4;
    auto F = [&](int i) { return (const float*)d_in[i]; };

    // Param leaf indices: jax tree-flatten order (sorted dict keys, lists in order).
    const int C1B0 = 2, C1B1 = 14;
    const int C2B0 = 26, C2B1 = 38, C2D = 50;   // bn b,g,m,v @50..53, w @54
    const int C3B0 = 55, C3B1 = 67, C3D = 79;   // w @83
    const int C4B0 = 84, C4B1 = 96, C4D = 108;  // w @112
    const int CIN_BN = 113, CIN_W = 117;        // conv_input
    const int COUT_BN = 118, COUT_W = 122;      // conv_out

    Bump ws{(char*)d_ws, 0};

    const int D0 = 41, H0 = 200, W0 = 200;
    const int D1 = 21, H1 = 100, W1 = 100;
    const int D2 = 11, H2 = 50,  W2 = 50;
    const int D3 = 5,  H3 = 25,  W3 = 25;
    const int D4 = 2;
    long M0 = (long)D0 * H0 * W0, M1 = (long)D1 * H1 * W1;
    long M2 = (long)D2 * H2 * W2, M3 = (long)D3 * H3 * W3, M4 = (long)D4 * H3 * W3;

    // 64B zero page for branchless OOB loads
    u16* zpad = (u16*)ws.alloc(64);

    // activation buffers (bf16) + masks
    u16* g0  = (u16*)ws.alloc(M0 * 16 * 2);
    u16* f0a = (u16*)ws.alloc(M0 * 16 * 2);
    u16* f0b = (u16*)ws.alloc(M0 * 16 * 2);
    u16* mk0 = (u16*)ws.alloc(M0 * 2);
    u16* f1a = (u16*)ws.alloc(M1 * 32 * 2);
    u16* f1b = (u16*)ws.alloc(M1 * 32 * 2);
    u16* f1c = (u16*)ws.alloc(M1 * 32 * 2);
    u16* mk1 = (u16*)ws.alloc(M1 * 2);
    u16* f2a = (u16*)ws.alloc(M2 * 64 * 2);
    u16* f2b = (u16*)ws.alloc(M2 * 64 * 2);
    u16* f2c = (u16*)ws.alloc(M2 * 64 * 2);
    u16* mk2 = (u16*)ws.alloc(M2 * 2);
    u16* f3a = (u16*)ws.alloc(M3 * 128 * 2);
    u16* f3b = (u16*)ws.alloc(M3 * 128 * 2);
    u16* f3c = (u16*)ws.alloc(M3 * 128 * 2);
    u16* mk3 = (u16*)ws.alloc(M3 * 2);
    u16* mk4 = (u16*)ws.alloc(M4 * 2);

    // packed weights + folded BN
    PC pci   = make_pc(ws, 27, 4, 16, 16);
    PC c1[4] = { make_pc(ws,27,16,16,16), make_pc(ws,27,16,16,16),
                 make_pc(ws,27,16,16,16), make_pc(ws,27,16,16,16) };
    PC p2d   = make_pc(ws, 27, 16, 16, 32);
    PC c2[4] = { make_pc(ws,27,32,32,32), make_pc(ws,27,32,32,32),
                 make_pc(ws,27,32,32,32), make_pc(ws,27,32,32,32) };
    PC p3d   = make_pc(ws, 27, 32, 32, 64);
    PC c3[4] = { make_pc(ws,27,64,64,64), make_pc(ws,27,64,64,64),
                 make_pc(ws,27,64,64,64), make_pc(ws,27,64,64,64) };
    PC p4d   = make_pc(ws, 27, 64, 64, 128);
    PC c4[4] = { make_pc(ws,27,128,128,128), make_pc(ws,27,128,128,128),
                 make_pc(ws,27,128,128,128), make_pc(ws,27,128,128,128) };
    PC pco   = make_pc(ws, 3, 128, 128, 128);

    // ---- zero page + zero/scatter input grid (re-done every call) ----
    k_zero16<<<1, 64, 0, stream>>>((uint4*)zpad, 4);
    k_zero16<<<(int)((M0 * 2 + 255) / 256), 256, 0, stream>>>((uint4*)g0, M0 * 2);
    k_zero16<<<(int)((M0 / 8 + 255) / 256), 256, 0, stream>>>((uint4*)mk0, M0 / 8);
    k_scatter<<<(N + 255) / 256, 256, 0, stream>>>(feat, coords, N, g0, mk0, H0, W0, 16);

    // ---- pack weights / fold BN ----
    prep_pc(pci, F(CIN_W), nullptr, F(CIN_BN), F(CIN_BN+1), F(CIN_BN+2), F(CIN_BN+3), stream);
    int bbase[2] = { C1B0, C1B1 };
    for (int b = 0; b < 2; ++b) {
        int B = bbase[b];
        prep_pc(c1[b*2+0], F(B+10), F(B+0), F(B+2), F(B+3), F(B+4), F(B+5), stream);
        prep_pc(c1[b*2+1], F(B+11), F(B+1), F(B+6), F(B+7), F(B+8), F(B+9), stream);
    }
    prep_pc(p2d, F(C2D+4), nullptr, F(C2D), F(C2D+1), F(C2D+2), F(C2D+3), stream);
    int b2b[2] = { C2B0, C2B1 };
    for (int b = 0; b < 2; ++b) {
        int B = b2b[b];
        prep_pc(c2[b*2+0], F(B+10), F(B+0), F(B+2), F(B+3), F(B+4), F(B+5), stream);
        prep_pc(c2[b*2+1], F(B+11), F(B+1), F(B+6), F(B+7), F(B+8), F(B+9), stream);
    }
    prep_pc(p3d, F(C3D+4), nullptr, F(C3D), F(C3D+1), F(C3D+2), F(C3D+3), stream);
    int b3b[2] = { C3B0, C3B1 };
    for (int b = 0; b < 2; ++b) {
        int B = b3b[b];
        prep_pc(c3[b*2+0], F(B+10), F(B+0), F(B+2), F(B+3), F(B+4), F(B+5), stream);
        prep_pc(c3[b*2+1], F(B+11), F(B+1), F(B+6), F(B+7), F(B+8), F(B+9), stream);
    }
    prep_pc(p4d, F(C4D+4), nullptr, F(C4D), F(C4D+1), F(C4D+2), F(C4D+3), stream);
    int b4b[2] = { C4B0, C4B1 };
    for (int b = 0; b < 2; ++b) {
        int B = b4b[b];
        prep_pc(c4[b*2+0], F(B+10), F(B+0), F(B+2), F(B+3), F(B+4), F(B+5), stream);
        prep_pc(c4[b*2+1], F(B+11), F(B+1), F(B+6), F(B+7), F(B+8), F(B+9), stream);
    }
    prep_pc(pco, F(COUT_W), nullptr, F(COUT_BN), F(COUT_BN+1), F(COUT_BN+2), F(COUT_BN+3), stream);

    // ---- network ----
    run_conv(g0, pci, mk0, nullptr, zpad, f0a, nullptr, D0,H0,W0, D0,H0,W0, 3,3,3, 1,1,1, 1,1,1, 1, stream);
    // conv1 block 0 (x = f0a)
    run_conv(f0a, c1[0], mk0, nullptr, zpad, f0b, nullptr, D0,H0,W0, D0,H0,W0, 3,3,3, 1,1,1, 1,1,1, 1, stream);
    run_conv(f0b, c1[1], mk0, f0a,    zpad, g0,  nullptr, D0,H0,W0, D0,H0,W0, 3,3,3, 1,1,1, 1,1,1, 1, stream);
    // conv1 block 1 (x = g0)
    run_conv(g0,  c1[2], mk0, nullptr, zpad, f0b, nullptr, D0,H0,W0, D0,H0,W0, 3,3,3, 1,1,1, 1,1,1, 1, stream);
    run_conv(f0b, c1[3], mk0, g0,     zpad, f0a, nullptr, D0,H0,W0, D0,H0,W0, 3,3,3, 1,1,1, 1,1,1, 1, stream);

    // stage 2
    k_maskdown<<<(int)((M1 + 255) / 256), 256, 0, stream>>>(mk0, mk1, D0,H0,W0, D1,H1,W1, 3,3,3, 2,2,2, 1,1,1);
    run_conv(f0a, p2d, mk1, nullptr, zpad, f1a, nullptr, D0,H0,W0, D1,H1,W1, 3,3,3, 2,2,2, 1,1,1, 1, stream);
    run_conv(f1a, c2[0], mk1, nullptr, zpad, f1b, nullptr, D1,H1,W1, D1,H1,W1, 3,3,3, 1,1,1, 1,1,1, 1, stream);
    run_conv(f1b, c2[1], mk1, f1a,    zpad, f1c, nullptr, D1,H1,W1, D1,H1,W1, 3,3,3, 1,1,1, 1,1,1, 1, stream);
    run_conv(f1c, c2[2], mk1, nullptr, zpad, f1b, nullptr, D1,H1,W1, D1,H1,W1, 3,3,3, 1,1,1, 1,1,1, 1, stream);
    run_conv(f1b, c2[3], mk1, f1c,    zpad, f1a, nullptr, D1,H1,W1, D1,H1,W1, 3,3,3, 1,1,1, 1,1,1, 1, stream);

    // stage 3
    k_maskdown<<<(int)((M2 + 255) / 256), 256, 0, stream>>>(mk1, mk2, D1,H1,W1, D2,H2,W2, 3,3,3, 2,2,2, 1,1,1);
    run_conv(f1a, p3d, mk2, nullptr, zpad, f2a, nullptr, D1,H1,W1, D2,H2,W2, 3,3,3, 2,2,2, 1,1,1, 1, stream);
    run_conv(f2a, c3[0], mk2, nullptr, zpad, f2b, nullptr, D2,H2,W2, D2,H2,W2, 3,3,3, 1,1,1, 1,1,1, 1, stream);
    run_conv(f2b, c3[1], mk2, f2a,    zpad, f2c, nullptr, D2,H2,W2, D2,H2,W2, 3,3,3, 1,1,1, 1,1,1, 1, stream);
    run_conv(f2c, c3[2], mk2, nullptr, zpad, f2b, nullptr, D2,H2,W2, D2,H2,W2, 3,3,3, 1,1,1, 1,1,1, 1, stream);
    run_conv(f2b, c3[3], mk2, f2c,    zpad, f2a, nullptr, D2,H2,W2, D2,H2,W2, 3,3,3, 1,1,1, 1,1,1, 1, stream);

    // stage 4 (pad D=0)
    k_maskdown<<<(int)((M3 + 255) / 256), 256, 0, stream>>>(mk2, mk3, D2,H2,W2, D3,H3,W3, 3,3,3, 2,2,2, 0,1,1);
    run_conv(f2a, p4d, mk3, nullptr, zpad, f3a, nullptr, D2,H2,W2, D3,H3,W3, 3,3,3, 2,2,2, 0,1,1, 1, stream);
    run_conv(f3a, c4[0], mk3, nullptr, zpad, f3b, nullptr, D3,H3,W3, D3,H3,W3, 3,3,3, 1,1,1, 1,1,1, 1, stream);
    run_conv(f3b, c4[1], mk3, f3a,    zpad, f3c, nullptr, D3,H3,W3, D3,H3,W3, 3,3,3, 1,1,1, 1,1,1, 1, stream);
    run_conv(f3c, c4[2], mk3, nullptr, zpad, f3b, nullptr, D3,H3,W3, D3,H3,W3, 3,3,3, 1,1,1, 1,1,1, 1, stream);
    run_conv(f3b, c4[3], mk3, f3c,    zpad, f3a, nullptr, D3,H3,W3, D3,H3,W3, 3,3,3, 1,1,1, 1,1,1, 1, stream);

    // conv_out: bn(conv k=(3,1,1), s=(2,1,1), p=0) * mo, fp32 out
    k_maskdown<<<(int)((M4 + 255) / 256), 256, 0, stream>>>(mk3, mk4, D3,H3,W3, D4,H3,W3, 3,1,1, 2,1,1, 0,0,0);
    run_conv(f3a, pco, mk4, nullptr, zpad, nullptr, (float*)d_out,
             D3,H3,W3, D4,H3,W3, 3,1,1, 2,1,1, 0,0,0, 0, stream);
}
